// PLIKS_11218454577413
// MI455X (gfx1250) — matrix operand order repaired and checked by reading, not compiled, not run on hardware
//
#include <hip/hip_runtime.h>
#include <math.h>

// ---------------- problem constants ----------------
#define BN    64        // batches
#define KN    24        // parts
#define NN    934       // points
#define COLS  154       // 6*K + 10 real columns
#define CP    160       // padded columns (10 WMMA tiles of 16)
#define PB    16        // points per row-block
#define RB    32        // rows per block (x-rows + y-rows)
#define NKB   59        // ceil(934/16)
#define SPLITS 4        // row-splits per batch
#define NTILE 55        // upper-tri 16x16 tiles of 160x160
#define LAM2  (0.1111111111111111f)   // (1/3)^2, lam = exp(log(1/3)*1.0)

typedef __attribute__((ext_vector_type(16))) __bf16 v16bf;
typedef __attribute__((ext_vector_type(8)))  float  v8f;

// fp32 -> bf16 round-to-nearest-even (bits)
static __device__ inline unsigned short f2bf(float f) {
    unsigned u = __float_as_uint(f);
    unsigned r = (u + 0x7FFFu + ((u >> 16) & 1u)) >> 16;
    return (unsigned short)r;
}
static __device__ inline float bf2f(unsigned short h) {
    return __uint_as_float(((unsigned)h) << 16);
}
static __device__ inline __bf16 bfbits(unsigned short h) {
    return __builtin_bit_cast(__bf16, h);
}
// upper-triangular tile index -> (ci, cj), ci <= cj, 10x10 tile grid
static __device__ inline void tile_from(int t, int& ci, int& cj) {
    int i = 0;
    while (t >= 10 - i) { t -= 10 - i; ++i; }
    ci = i; cj = i + t;
}

// ---------------- kernel 0: init G and g ----------------
__global__ __launch_bounds__(256) void pliks_init(const float* __restrict__ pbeta,
                                                  float* __restrict__ G,
                                                  float* __restrict__ g) {
    const long long GTOT = (long long)BN * CP * CP;
    long long idx = (long long)blockIdx.x * 256 + threadIdx.x;
    if (idx < GTOT) {
        int rc = (int)(idx % (CP * CP));
        int r = rc / CP, c = rc % CP;
        G[idx] = (r == c && r >= 144 && r < 154) ? LAM2 : 0.0f;
    } else if (idx < GTOT + (long long)BN * CP) {
        long long j = idx - GTOT;
        int b = (int)(j / CP), c = (int)(j % CP);
        g[j] = (c >= 144 && c < 154) ? LAM2 * pbeta[b * 10 + (c - 144)] : 0.0f;
    }
}

// ---------------- kernel 1: streaming Gram build with bf16 WMMA ----------------
__global__ __launch_bounds__(256) void pliks_gram(
    const float* __restrict__ uv, const float* __restrict__ xm,
    const float* __restrict__ bs, const float* __restrict__ blend,
    const float* __restrict__ W,  const float* __restrict__ m00,
    const float* __restrict__ m11, const float* __restrict__ m02,
    const float* __restrict__ m12,
    float* __restrict__ G, float* __restrict__ g) {

    // A-block stored column-major: S[col][row], rows contiguous -> fragment
    // loads are contiguous 16-element (32B) runs in LDS.
    __shared__ unsigned short Shi[CP][RB];
    __shared__ unsigned short Slo[CP][RB];
    __shared__ float bvec[RB];

    const int b     = blockIdx.x / SPLITS;
    const int split = blockIdx.x % SPLITS;
    const int tid   = threadIdx.x;
    const int lane  = tid & 31;
    const int wave  = tid >> 5;

    const float f0 = m00[b], f1 = m11[b];
    const float c02 = m02[b], c12 = m12[b];

    v8f acc[7];
    for (int i = 0; i < 7; ++i)
        for (int e = 0; e < 8; ++e) acc[i][e] = 0.0f;
    float gacc = 0.0f;

    // ---- hoist per-wave tile assignment + LDS fragment bases out of the
    //      k-block loop (loop-invariant; keeps hot loop to addr+ds+wmma) ----
    const int m    = lane & 15;          // A-fragment row / B-fragment col
    const int half = lane >> 4;          // lane group
    const int arun = 16 * half;          // A-fragment K-run start within row
    int  tci[7], tcj[7];
    int  aoffs[7], boffs[7];             // element offsets into Shi/Slo
    bool tval[7];
    for (int lt = 0; lt < 7; ++lt) {
        const int t = lt * 8 + wave;     // wave-uniform
        tval[lt] = (t < NTILE);
        int ci = 0, cj = 0;
        if (t < NTILE) tile_from(t, ci, cj);
        tci[lt] = ci; tcj[lt] = cj;
        aoffs[lt] = (16 * ci + m) * RB;
        boffs[lt] = (16 * cj + m) * RB + 16 * half;
    }
    const unsigned short* SH = &Shi[0][0];
    const unsigned short* SL = &Slo[0][0];

    for (int kb = split; kb < NKB; kb += SPLITS) {
        const int n0 = kb * PB;

        // ---- phase A: 6K pose columns (16 points x 24 parts = 384 tasks) ----
        for (int task = tid; task < PB * KN; task += 256) {
            const int p = task & 15;
            const int k = task >> 4;
            const int n = n0 + p;
            float xv[6] = {0,0,0,0,0,0}, yv[6] = {0,0,0,0,0,0};
            if (n < NN) {
                const float w   = W[b * NN + n];
                const float dxu = c02 - uv[((size_t)b * NN + n) * 2 + 0];
                const float dyv = c12 - uv[((size_t)b * NN + n) * 2 + 1];
                const size_t xb = (((size_t)b * KN + k) * NN + n) * 3;
                const float x0 = xm[xb], x1m = xm[xb + 1], x2m = xm[xb + 2];
                const float bl = blend[(size_t)n * KN + k];
                xv[0] = w * (dxu * x1m);
                xv[1] = w * (f0 * x2m - dxu * x0);
                xv[2] = w * (-f0 * x1m);
                xv[3] = w * (bl * f0);
                xv[4] = 0.0f;
                xv[5] = w * (bl * dxu);
                yv[0] = w * (-f1 * x2m + dyv * x1m);
                yv[1] = w * (-dyv * x0);
                yv[2] = w * (f1 * x0);
                yv[3] = 0.0f;
                yv[4] = w * (bl * f1);
                yv[5] = w * (bl * dyv);
            }
            const int col = k * 6;
            for (int c = 0; c < 6; ++c) {
                unsigned short hb = f2bf(xv[c]);
                Shi[col + c][p] = hb;
                Slo[col + c][p] = f2bf(xv[c] - bf2f(hb));
                hb = f2bf(yv[c]);
                Shi[col + c][16 + p] = hb;
                Slo[col + c][16 + p] = f2bf(yv[c] - bf2f(hb));
            }
        }

        // ---- phase B: shape columns (b_s_ reductions over K), rhs b, padding ----
        {
            const int p = tid & 15;
            const int u = tid >> 4;           // 0..15
            const int n = n0 + p;
            if (u < 10) {
                float sx = 0.0f, sy = 0.0f;
                if (n < NN) {
                    const float w   = W[b * NN + n];
                    const float dxu = c02 - uv[((size_t)b * NN + n) * 2 + 0];
                    const float dyv = c12 - uv[((size_t)b * NN + n) * 2 + 1];
                    for (int k = 0; k < KN; ++k) {
                        const size_t base =
                            ((((size_t)b * KN + k) * NN + n) * 3) * 10 + u;
                        __builtin_prefetch(&bs[base + (size_t)NN * 30], 0, 1);
                        const float b0 = bs[base];
                        const float b1 = bs[base + 10];
                        const float b2 = bs[base + 20];
                        sx += f0 * b0 + dxu * b2;
                        sy += f1 * b1 + dyv * b2;
                    }
                    sx *= w; sy *= w;
                }
                unsigned short hb = f2bf(sx);
                Shi[144 + u][p] = hb;
                Slo[144 + u][p] = f2bf(sx - bf2f(hb));
                hb = f2bf(sy);
                Shi[144 + u][16 + p] = hb;
                Slo[144 + u][16 + p] = f2bf(sy - bf2f(hb));
            } else if (u == 10 || u == 11) {
                float s = 0.0f;
                if (n < NN) {
                    const float w   = W[b * NN + n];
                    const float dxu = c02 - uv[((size_t)b * NN + n) * 2 + 0];
                    const float dyv = c12 - uv[((size_t)b * NN + n) * 2 + 1];
                    for (int k = 0; k < KN; ++k) {
                        const size_t xb = (((size_t)b * KN + k) * NN + n) * 3;
                        const float x0 = xm[xb], x1m = xm[xb + 1], x2m = xm[xb + 2];
                        s += (u == 10) ? (-f0 * x0 - dxu * x2m)
                                       : (-f1 * x1m - dyv * x2m);
                    }
                    s *= w;
                }
                bvec[(u == 10 ? 0 : 16) + p] = s;
            } else if (u <= 14) {             // zero pad columns 154..159
                const int c0 = 154 + (u - 12) * 2;
                for (int c = c0; c < c0 + 2; ++c) {
                    Shi[c][p] = 0; Slo[c][p] = 0;
                    Shi[c][16 + p] = 0; Slo[c][16 + p] = 0;
                }
            }
        }
        __syncthreads();

        // ---- WMMA phase: G_tile += (A-block)^T (A-block), split bf16 ----
        for (int lt = 0; lt < 7; ++lt) {
            if (tval[lt]) {                   // wave-uniform -> EXEC all ones
                const unsigned short* qh = SH + aoffs[lt];
                const unsigned short* ql = SL + aoffs[lt];
                v16bf ahi, alo;
                for (int e = 0; e < 8; ++e) {
                    ahi[e]     = bfbits(qh[arun + e]);
                    alo[e]     = bfbits(ql[arun + e]);
                    ahi[8 + e] = bfbits(qh[8 + arun + e]);
                    alo[8 + e] = bfbits(ql[8 + arun + e]);
                }
                const unsigned short* ph = SH + boffs[lt];
                const unsigned short* pl = SL + boffs[lt];
                v16bf bhi, blo;
                for (int h = 0; h < 16; ++h) {
                    bhi[h] = bfbits(ph[h]);
                    blo[h] = bfbits(pl[h]);
                }
                // v*v ~ hi*hi + hi*lo + lo*hi  (fp32-accurate Gram)
                acc[lt] = __builtin_amdgcn_wmma_f32_16x16x32_bf16(
                    false, ahi, false, bhi, (short)0, acc[lt], false, false);
                acc[lt] = __builtin_amdgcn_wmma_f32_16x16x32_bf16(
                    false, ahi, false, blo, (short)0, acc[lt], false, false);
                acc[lt] = __builtin_amdgcn_wmma_f32_16x16x32_bf16(
                    false, alo, false, bhi, (short)0, acc[lt], false, false);
            }
        }

        // ---- g += A^T b on VALU ----
        if (tid < CP) {
            float s = 0.0f;
            for (int r = 0; r < RB; ++r)
                s += (bf2f(Shi[tid][r]) + bf2f(Slo[tid][r])) * bvec[r];
            gacc += s;
        }
        __syncthreads();
    }

    // ---- writeback: atomic accumulate Gram tiles (+ symmetric mirror) ----
    for (int lt = 0; lt < 7; ++lt) {
        if (tval[lt]) {
            const int ci = tci[lt], cj = tcj[lt];
            const int mh = 8 * half;
            float* Gb = G + (size_t)b * CP * CP;
            for (int r = 0; r < 8; ++r) {
                const int row = 16 * ci + mh + r;
                const int col = 16 * cj + m;
                const float v = acc[lt][r];
                atomicAdd(&Gb[(size_t)row * CP + col], v);
                if (ci != cj) atomicAdd(&Gb[(size_t)col * CP + row], v);
            }
        }
    }
    if (tid < CP) atomicAdd(&g[(size_t)b * CP + tid], gacc);
}

// ---------------- kernel 2: per-batch Cholesky solve (154x154 SPD) ----------------
__global__ __launch_bounds__(256) void pliks_chol(const float* __restrict__ G,
                                                  const float* __restrict__ g,
                                                  float* __restrict__ sol) {
    __shared__ float L[COLS * (COLS + 1) / 2];   // packed lower triangle, 47.7 KB
    __shared__ float rhs[COLS];
    __shared__ float s_bc;
    const int b = blockIdx.x, tid = threadIdx.x;

    for (int i = tid; i < COLS; i += 256) {
        const float* row = &G[(size_t)b * CP * CP + (size_t)i * CP];
        float* dst = &L[i * (i + 1) / 2];
        for (int j = 0; j <= i; ++j) dst[j] = row[j];
    }
    for (int i = tid; i < COLS; i += 256) rhs[i] = g[(size_t)b * CP + i];
    __syncthreads();

    // factor: G = L L^T
    for (int j = 0; j < COLS; ++j) {
        if (tid == 0) {
            const float d = sqrtf(fmaxf(L[j * (j + 1) / 2 + j], 1e-20f));
            L[j * (j + 1) / 2 + j] = d;
            s_bc = 1.0f / d;
        }
        __syncthreads();
        const float dinv = s_bc;
        for (int i = j + 1 + tid; i < COLS; i += 256)
            L[i * (i + 1) / 2 + j] *= dinv;
        __syncthreads();
        for (int i = j + 1 + tid; i < COLS; i += 256) {
            const float lij = L[i * (i + 1) / 2 + j];
            float* Li = &L[i * (i + 1) / 2];
            for (int r = j + 1; r <= i; ++r)
                Li[r] -= lij * L[r * (r + 1) / 2 + j];
        }
        __syncthreads();
    }
    // forward: L y = g
    for (int j = 0; j < COLS; ++j) {
        if (tid == 0) rhs[j] /= L[j * (j + 1) / 2 + j];
        __syncthreads();
        const float yj = rhs[j];
        for (int i = j + 1 + tid; i < COLS; i += 256)
            rhs[i] -= L[i * (i + 1) / 2 + j] * yj;
        __syncthreads();
    }
    // backward: L^T x = y
    for (int j = COLS - 1; j >= 0; --j) {
        if (tid == 0) rhs[j] /= L[j * (j + 1) / 2 + j];
        __syncthreads();
        const float xj = rhs[j];
        for (int i = tid; i < j; i += 256)
            rhs[i] -= L[j * (j + 1) / 2 + i] * xj;
        __syncthreads();
    }
    for (int i = tid; i < COLS; i += 256) sol[(size_t)b * CP + i] = rhs[i];
}

// ---------------- kernel 3: Rodrigues + 4x4 compose + scatter outputs ----------------
__global__ __launch_bounds__(256) void pliks_final(const float* __restrict__ sol,
                                                   const float* __restrict__ Rlist,
                                                   float* __restrict__ out) {
    const int t = blockIdx.x * 256 + threadIdx.x;
    if (t >= BN * KN) return;
    const int b = t / KN, k = t % KN;
    const float* s = &sol[(size_t)b * CP + k * 6];

    const float rx = s[0], ry = s[1], rz = s[2];
    float theta = sqrtf(rx * rx + ry * ry + rz * rz);
    theta = fmaxf(theta, 1e-8f);
    const float ax = rx / theta, ay = ry / theta, az = rz / theta;
    const float c = cosf(theta), sn = sinf(theta), mc = 1.0f - c;
    float R[3][3];
    R[0][0] = c + mc * ax * ax;        R[0][1] = -sn * az + mc * ax * ay; R[0][2] = sn * ay + mc * ax * az;
    R[1][0] = sn * az + mc * ay * ax;  R[1][1] = c + mc * ay * ay;        R[1][2] = -sn * ax + mc * ay * az;
    R[2][0] = -sn * ay + mc * az * ax; R[2][1] = sn * ax + mc * az * ay;  R[2][2] = c + mc * az * az;

    float M4[4][4];
    for (int i = 0; i < 3; ++i) {
        for (int j = 0; j < 3; ++j) M4[i][j] = R[i][j];
        M4[i][3] = 0.0f;
    }
    M4[3][0] = 0.0f; M4[3][1] = 0.0f; M4[3][2] = 0.0f; M4[3][3] = 1.0f;

    const float* V = &Rlist[((size_t)b * KN + k) * 16];
    float* rc = out + (size_t)(b * KN + k) * 16;                    // r_curr
    for (int i = 0; i < 4; ++i)
        for (int j = 0; j < 4; ++j) {
            float acc = 0.0f;
            for (int mm = 0; mm < 4; ++mm) acc += M4[i][mm] * V[mm * 4 + j];
            rc[i * 4 + j] = acc;
        }
    float* tp = out + BN * KN * 16 + (size_t)(b * KN + k) * 3;      // t_pred
    tp[0] = s[3]; tp[1] = s[4]; tp[2] = s[5];
    if (k < 10)                                                     // beta
        out[BN * KN * 16 + BN * KN * 3 + b * 10 + k] =
            sol[(size_t)b * CP + 144 + k];
    float* rp = out + BN * KN * 16 + BN * KN * 3 + BN * 10 +        // r_pred
                (size_t)(b * KN + k) * 9;
    for (int i = 0; i < 3; ++i)
        for (int j = 0; j < 3; ++j) rp[i * 3 + j] = R[i][j];
}

extern "C" void kernel_launch(void* const* d_in, const int* in_sizes, int n_in,
                              void* d_out, int out_size, void* d_ws, size_t ws_size,
                              hipStream_t stream) {
    const float* uv    = (const float*)d_in[0];
    const float* xm    = (const float*)d_in[1];
    const float* bs    = (const float*)d_in[2];
    const float* blend = (const float*)d_in[3];
    const float* W     = (const float*)d_in[4];
    const float* m00   = (const float*)d_in[5];
    const float* m11   = (const float*)d_in[6];
    const float* m02   = (const float*)d_in[7];
    const float* m12   = (const float*)d_in[8];
    const float* pbeta = (const float*)d_in[9];
    const float* Rlist = (const float*)d_in[10];

    float* G   = (float*)d_ws;                          // B*160*160
    float* g   = G + (size_t)BN * CP * CP;              // B*160
    float* sol = g + (size_t)BN * CP;                   // B*160

    const long long initN = (long long)BN * CP * CP + (long long)BN * CP;
    pliks_init<<<(int)((initN + 255) / 256), 256, 0, stream>>>(pbeta, G, g);
    pliks_gram<<<BN * SPLITS, 256, 0, stream>>>(uv, xm, bs, blend, W,
                                                m00, m11, m02, m12, G, g);
    pliks_chol<<<BN, 256, 0, stream>>>(G, g, sol);
    pliks_final<<<(BN * KN + 255) / 256, 256, 0, stream>>>(sol, Rlist,
                                                           (float*)d_out);
}
